// customRNN_59528246722699
// MI455X (gfx1250) — matrix operand order repaired and checked by reading, not compiled, not run on hardware
//
#include <hip/hip_runtime.h>

// ---------------------------------------------------------------------------
// customRNN for MI455X (gfx1250, wave32, WMMA)
//   1) hc = tanh(U @ Wi^T + bi)      : WMMA bf16 GEMM (fp32 accum), bf16 out
//      B staged in LDS via async global->LDS (double buffered)
//   2) cc = hc @ Wh_cur^T + bh       : same GEMM, f32 out
//   3) serial scan over 32768 steps  : 16-WG cluster, Wh_prev slab pulled into
//      LDS by the Tensor Data Mover, cluster barrier per step, WG0 softmax.
// ---------------------------------------------------------------------------

typedef __bf16 bf16;
typedef __attribute__((ext_vector_type(16))) __bf16 v16bf;
typedef __attribute__((ext_vector_type(8)))  __bf16 v8bf;
typedef __attribute__((ext_vector_type(8)))  float  v8f;
typedef __attribute__((ext_vector_type(4)))  unsigned int v4u;
typedef __attribute__((ext_vector_type(8)))  int v8i;
typedef __attribute__((ext_vector_type(4)))  int v4i;

#define T_STEPS 32768
#define INP 1024
#define HID 1024
#define OUT_N 7
#define NWG 16          // cluster width for the scan
#define ROWS 64         // hidden rows owned per scan workgroup

#if __has_builtin(__builtin_amdgcn_global_load_async_to_lds_b128) && \
    __has_builtin(__builtin_amdgcn_s_wait_asynccnt)
#define USE_ASYNC_LDS 1
#endif

// Typed address-space pointers for the async builtin (signature revealed by
// the compiler: v4i pointers). Materialized via integer casts (HIP-safe);
// LDS generic pointers carry the LDS byte offset in the low 32 bits.
#define TO_AS1_V4I(p) ((v4i __attribute__((address_space(1))) *)(unsigned long long)(p))
#define TO_AS3_V4I(p) ((v4i __attribute__((address_space(3))) *)(unsigned int)(unsigned long long)(p))

// 16B global -> LDS copy: async (ASYNCcnt) when available, else ds_store.
__device__ __forceinline__ void copy16_to_lds(void* lds_dst, const void* gsrc) {
#ifdef USE_ASYNC_LDS
    __builtin_amdgcn_global_load_async_to_lds_b128(TO_AS1_V4I(gsrc), TO_AS3_V4I(lds_dst), 0, 0);
#else
    *(v8bf*)lds_dst = *(const v8bf*)gsrc;
#endif
}
__device__ __forceinline__ void wait_lds_copies() {
#ifdef USE_ASYNC_LDS
    __builtin_amdgcn_s_wait_asynccnt(0);
#endif
}

// ----------------------------- fragment loads ------------------------------
// A (16x32 bf16, MxK), ISA 7.12.2: lanes 0-15 row M=lane hold K {0..7,16..23},
// lanes 16-31 row M=lane-16 hold K {8..15,24..31}; 2 K-values per VGPR.
__device__ __forceinline__ v16bf load_fragA_bf16(const bf16* p, int ld, int k0, int lane) {
    int r  = lane & 15;
    int hi = (lane >> 4) & 1;
    const bf16* base = p + (size_t)r * ld + k0 + hi * 8;
    v8bf a0 = *(const v8bf*)(base);
    v8bf a1 = *(const v8bf*)(base + 16);
    v16bf o;
#pragma unroll
    for (int i = 0; i < 8; ++i) { o[i] = a0[i]; o[i + 8] = a1[i]; }
    return o;
}

__device__ __forceinline__ v16bf load_fragA_f32(const float* p, int ld, int k0, int lane) {
    int r  = lane & 15;
    int hi = (lane >> 4) & 1;
    const float* base = p + (size_t)r * ld + k0 + hi * 8;
    v16bf o;
#pragma unroll
    for (int i = 0; i < 8; ++i) { o[i] = (bf16)base[i]; o[i + 8] = (bf16)base[i + 16]; }
    return o;
}

// B fragment read out of the LDS-staged tile (row-major [64][32] bf16):
// lane holds column n = nbase + (lane&15); lanes 0-15 K {0..7,16..23}, 16-31 K {8..15,24..31}.
__device__ __forceinline__ v16bf load_fragB_lds(const bf16* tile, int nbase, int lane) {
    int n  = nbase + (lane & 15);
    int hi = (lane >> 4) & 1;
    const bf16* base = tile + n * 32 + hi * 8;
    v8bf b0 = *(const v8bf*)(base);      // ds_load_b128
    v8bf b1 = *(const v8bf*)(base + 16);
    v16bf o;
#pragma unroll
    for (int i = 0; i < 8; ++i) { o[i] = b0[i]; o[i + 8] = b1[i]; }
    return o;
}

// ----------------------------- GEMM (WMMA) ---------------------------------
// C(MxN) = act(A(MxK) @ B(NxK)^T + bias). Wave tile 16x64 (4 wmma / K step),
// block = 8 waves -> 128x64. B tile (64x32 = 4KB) double-buffered in LDS,
// filled by async global->LDS copies that overlap the WMMAs of the prior step.
template <bool A_F32, bool TANH_OUT, bool OUT_BF16>
__global__ __launch_bounds__(256) void gemm_wmma_kernel(
    const void* __restrict__ Ap, const bf16* __restrict__ B,
    const float* __restrict__ bias, void* __restrict__ Out,
    int M, int N, int K) {
    __shared__ bf16 Bs[2][64 * 32];     // 8 KB

    const int tid  = threadIdx.x;
    const int lane = tid & 31;
    const int wave = tid >> 5;
    const int m0 = (blockIdx.x * 8 + wave) * 16;
    const int n0 = blockIdx.y * 64;

    const float* Af = (const float*)Ap;
    const bf16*  Ab = (const bf16*)Ap;

    // staging: 256 threads x 16B = 4KB tile; thread -> (row sn, K-chunk sc)
    const int sn = tid >> 2;
    const int sc = (tid & 3) * 8;
    auto stage = [&](int buf, int k0) {
        copy16_to_lds(&Bs[buf][sn * 32 + sc], B + (size_t)(n0 + sn) * K + k0 + sc);
    };

    stage(0, 0);
    wait_lds_copies();
    __syncthreads();

    v8f c[4] = {};
    for (int k0 = 0; k0 < K; k0 += 32) {
        const int cur = (k0 >> 5) & 1;
        if (k0 + 32 < K) {
            stage(cur ^ 1, k0 + 32);    // overlaps with this step's WMMAs
            if (A_F32) __builtin_prefetch(Af + (size_t)(m0 + (lane & 15)) * K + k0 + 32, 0, 0);
            else       __builtin_prefetch(Ab + (size_t)(m0 + (lane & 15)) * K + k0 + 32, 0, 0);
        }
        v16bf a = A_F32 ? load_fragA_f32(Af + (size_t)m0 * K, K, k0, lane)
                        : load_fragA_bf16(Ab + (size_t)m0 * K, K, k0, lane);
#pragma unroll
        for (int j = 0; j < 4; ++j) {
            v16bf b = load_fragB_lds(Bs[cur], j * 16, lane);
            c[j] = __builtin_amdgcn_wmma_f32_16x16x32_bf16(
                false, a, false, b, (short)0, c[j], false, false);
        }
        wait_lds_copies();
        __syncthreads();
    }

    // C/D layout: VGPR v, lanes 0-15 -> M=v, N=lane; lanes 16-31 -> M=8+v, N=lane-16
    const int col   = lane & 15;
    const int rbase = (lane >> 4) * 8;
#pragma unroll
    for (int j = 0; j < 4; ++j) {
        int n = n0 + j * 16 + col;
        float bv = bias[n];
#pragma unroll
        for (int v = 0; v < 8; ++v) {
            int m = m0 + rbase + v;
            float val = c[j][v] + bv;
            if (TANH_OUT) val = tanhf(val);
            if (OUT_BF16) ((bf16*)Out)[(size_t)m * N + n] = (bf16)val;
            else          ((float*)Out)[(size_t)m * N + n] = val;
        }
    }
}

// --------------------------- weight conversion -----------------------------
__global__ void cvt_f32_to_bf16_kernel(const float* __restrict__ src,
                                       bf16* __restrict__ dst,
                                       int rows, int cols, int src_ld, int src_off) {
    size_t i = (size_t)blockIdx.x * 256 + threadIdx.x;
    if (i >= (size_t)rows * cols) return;
    int r = (int)(i / cols), c = (int)(i % cols);
    dst[i] = (bf16)src[(size_t)r * src_ld + src_off + c];
}

__global__ void scan_init_kernel(float* __restrict__ hbuf, int* __restrict__ ctr) {
    int i = threadIdx.x;
    for (int k = i; k < 2 * HID; k += 256) hbuf[k] = 0.0f;
    if (i == 0) *ctr = 0;
}

// ------------------------------- scan kernel -------------------------------
__global__ __launch_bounds__(256) void rnn_scan_kernel(
    const bf16* __restrict__ Whp, const float* __restrict__ cc,
    const float* __restrict__ Wo, const float* __restrict__ bo,
    float* __restrict__ hbuf, int* __restrict__ ctr, float* __restrict__ out) {
    extern __shared__ char smem[];
    bf16*  slab = (bf16*)smem;                              // 64*1024*2   = 128 KB
    float* hs   = (float*)(smem + ROWS * HID * 2);          // 1024 f32
    float* part = hs + HID;                                 // 256 f32
    float* wos  = part + 256;                               // 7*1024 f32
    float* lg   = wos + OUT_N * HID;                        // 8 f32
    float* acc  = lg + 8;                                   // 8 f32

    const int tid = threadIdx.x;
    const int wg  = blockIdx.x;
    const bf16* src = Whp + (size_t)wg * ROWS * HID;

    // ---- Stage this WG's Wh_prev slab (64x1024 bf16 tile) into LDS ----
#if __has_builtin(__builtin_amdgcn_tensor_load_to_lds)
    if (tid == 0) {
        // D# per ISA ch.8: 2D tensor 1024x1024 bf16, tile 1024x64 at row wg*64.
        unsigned long long ga = (unsigned long long)src;        // tile global addr
        unsigned int lds_off = (unsigned int)(unsigned long long)slab;
        v4u g0;
        g0[0] = 1u;                                             // count=1 (valid)
        g0[1] = lds_off;                                        // lds_addr
        g0[2] = (unsigned int)ga;                               // global_addr[31:0]
        g0[3] = (unsigned int)((ga >> 32) & 0x01FFFFFFu)        // global_addr[56:32]
              | 0x80000000u;                                    // type=2 (image)
        v8i g1;
        g1[0] = (int)(1u << 16);          // wg_mask=0, data_size=1 (2 bytes)
        g1[1] = (int)(1024u << 16);       // tensor_dim0[15:0]=1024
        g1[2] = (int)(1024u << 16);       // tensor_dim0[31:16]=0 | tensor_dim1[15:0]=1024
        g1[3] = (int)(1024u << 16);       // tensor_dim1[31:16]=0 | tile_dim0=1024
        g1[4] = 64;                       // tile_dim1=64, tile_dim2=0
        g1[5] = 1024;                     // tensor_dim0_stride[31:0]=1024
        g1[6] = 0;                        // stride hi, tensor_dim1_stride lo
        g1[7] = 0;
        v4i gz = {0, 0, 0, 0};
#if __clang_major__ >= 23
        v8i gz8 = {0, 0, 0, 0, 0, 0, 0, 0};
        __builtin_amdgcn_tensor_load_to_lds(g0, g1, gz, gz, gz8, 0);
#else
        __builtin_amdgcn_tensor_load_to_lds(g0, g1, gz, gz, 0);
#endif
    }
    __builtin_amdgcn_s_wait_tensorcnt(0);
#else
    for (int i = tid; i < ROWS * HID / 8; i += 256)
        ((v8bf*)slab)[i] = ((const v8bf*)src)[i];
#endif
    if (wg == 0) {
        for (int i = tid; i < OUT_N * HID; i += 256) wos[i] = Wo[i];
        if (tid < 8) acc[tid] = 0.0f;
    }
    __syncthreads();

    const int row   = tid & (ROWS - 1);   // 0..63
    const int chunk = tid >> 6;           // 0..3 -> 256-wide K slice

    for (int t = 0; t < T_STEPS; ++t) {
        const int p = t & 1;
        if (t == 0) {
            if (tid < 256) ((float4*)hs)[tid] = make_float4(0.f, 0.f, 0.f, 0.f);
        } else {
            if (tid < 256) ((float4*)hs)[tid] = ((const float4*)(hbuf + p * HID))[tid];
        }
        __syncthreads();

        float s = 0.0f;
        const bf16*  wr = slab + row * HID + chunk * 256;
        const float* hv = hs + chunk * 256;
#pragma unroll 8
        for (int k = 0; k < 256; ++k) s = fmaf((float)wr[k], hv[k], s);
        part[tid] = s;
        __syncthreads();

        if (tid < ROWS) {
            float v = part[tid] + part[tid + 64] + part[tid + 128] + part[tid + 192];
            v += cc[(size_t)t * HID + wg * ROWS + tid];
            v = tanhf(v);
            hbuf[(1 - p) * HID + wg * ROWS + tid] = v;
        }
        __threadfence();
        __syncthreads();

        // Cluster sync (s_barrier_signal/-wait -3; NOP when not clustered)
        // + agent-scope atomic barrier fallback for plain dispatch.
        __builtin_amdgcn_s_cluster_barrier();
        if (tid == 0) {
            __hip_atomic_fetch_add(ctr, 1, __ATOMIC_ACQ_REL, __HIP_MEMORY_SCOPE_AGENT);
            const int target = (t + 1) * NWG;
            while (__hip_atomic_load(ctr, __ATOMIC_ACQUIRE, __HIP_MEMORY_SCOPE_AGENT) < target)
                __builtin_amdgcn_s_sleep(1);
        }
        __syncthreads();

        if (wg == 0) {
            const float* hn = hbuf + (1 - p) * HID;
            const int o = tid >> 5, l = tid & 31;
            if (o < OUT_N) {
                float ps = 0.0f;
                const float* wrow = wos + o * HID;
                for (int k = l; k < HID; k += 32) ps = fmaf(wrow[k], hn[k], ps);
#pragma unroll
                for (int off = 16; off > 0; off >>= 1) ps += __shfl_down(ps, off, 32);
                if (l == 0) lg[o] = ps + bo[o];
            }
            __syncthreads();
            if (tid == 0) {
                float mx = lg[0];
                for (int o2 = 1; o2 < OUT_N; ++o2) mx = fmaxf(mx, lg[o2]);
                float den = 0.0f, e[OUT_N];
                for (int o2 = 0; o2 < OUT_N; ++o2) { e[o2] = __expf(lg[o2] - mx); den += e[o2]; }
                const float inv = 1.0f / den;
                for (int o2 = 0; o2 < OUT_N; ++o2) acc[o2] += e[o2] * inv;
            }
            __syncthreads();
        }
    }
    if (wg == 0 && tid < OUT_N) out[tid] = acc[tid] * (1.0f / (float)T_STEPS);
}

// ------------------------------- launcher ----------------------------------
extern "C" void kernel_launch(void* const* d_in, const int* in_sizes, int n_in,
                              void* d_out, int out_size, void* d_ws, size_t ws_size,
                              hipStream_t stream) {
    const float* U  = (const float*)d_in[0];  // (T, INP)
    const float* Wi = (const float*)d_in[1];  // (HID, INP)
    const float* bi = (const float*)d_in[2];  // (HID,)
    const float* Wh = (const float*)d_in[3];  // (HID, 2*HID)
    const float* bh = (const float*)d_in[4];  // (HID,)
    const float* Wo = (const float*)d_in[5];  // (OUT, HID)
    const float* bo = (const float*)d_in[6];  // (OUT,)
    float* out = (float*)d_out;

    char* ws = (char*)d_ws;
    const size_t MB = 1ull << 20;
    bf16*  Wi_b  = (bf16*)(ws + 0 * MB);     //   2 MB
    bf16*  Whc_b = (bf16*)(ws + 2 * MB);     //   2 MB
    bf16*  Whp_b = (bf16*)(ws + 4 * MB);     //   2 MB
    bf16*  hc_b  = (bf16*)(ws + 6 * MB);     //  64 MB
    float* cc    = (float*)(ws + 70 * MB);   // 128 MB
    float* hbuf  = (float*)(ws + 198 * MB);  //   8 KB (ping-pong h)
    int*   ctr   = (int*)(ws + 198 * MB + 8192);
    if (ws_size < 199 * MB) return;          // insufficient scratch

    // 1) weights -> bf16 (Wh split into prev/cur halves, compacted)
    {
        int n = HID * INP, g = (n + 255) / 256;
        cvt_f32_to_bf16_kernel<<<g, 256, 0, stream>>>(Wi, Wi_b,  HID, INP, INP,     0);
        cvt_f32_to_bf16_kernel<<<g, 256, 0, stream>>>(Wh, Whp_b, HID, HID, 2 * HID, 0);
        cvt_f32_to_bf16_kernel<<<g, 256, 0, stream>>>(Wh, Whc_b, HID, HID, 2 * HID, HID);
    }

    // 2) hc = tanh(U @ Wi^T + bi)  -> bf16
    dim3 g1(T_STEPS / 128, HID / 64);
    gemm_wmma_kernel<true, true, true><<<g1, 256, 0, stream>>>(
        (const void*)U, Wi_b, bi, (void*)hc_b, T_STEPS, HID, INP);

    // 3) cc = hc @ Wh_cur^T + bh   -> f32
    gemm_wmma_kernel<false, false, false><<<g1, 256, 0, stream>>>(
        (const void*)hc_b, Whc_b, bh, (void*)cc, T_STEPS, HID, HID);

    // 4) scan state init + persistent cluster scan
    scan_init_kernel<<<1, 256, 0, stream>>>(hbuf, ctr);
    const size_t scan_lds = (size_t)ROWS * HID * 2      // Wh_prev slab
                          + HID * 4                     // h_prev
                          + 256 * 4                     // partials
                          + OUT_N * HID * 4             // Wo (WG0)
                          + 16 * 4;                     // logits + acc
    rnn_scan_kernel<<<NWG, 256, scan_lds, stream>>>(Whp_b, cc, Wo, bo, hbuf, ctr, out);
}
